// TorchLMHeadJSD_38225208934437
// MI455X (gfx1250) — matrix operand order repaired and checked by reading, not compiled, not run on hardware
//
#include <hip/hip_runtime.h>

#define BT      2048
#define HT      4096
#define HS      2048
#define VSZ     32000
#define TOKB    32                 // tokens per workgroup (2 M-tiles of 16)
#define VCHUNK  256                // vocab per workgroup (8 waves x 2 N-tiles)
#define NCHUNKS (VSZ / VCHUNK)     // 125
#define TBLKS   (BT / TOKB)        // 64
#define WAVES   8
#define THREADS 256
#define KSTEP   128                // K staged in LDS per barrier pair
#define LDSTR   (KSTEP + 8)        // padded LDS row stride (bf16 elems) -> bank-conflict-free

typedef __attribute__((ext_vector_type(16))) __bf16 v16bf;
typedef __attribute__((ext_vector_type(8)))  __bf16 v8bf;
typedef __attribute__((ext_vector_type(4)))  __bf16 v4bf;
typedef __attribute__((ext_vector_type(8)))  float  v8f;

// ---- B fragment: lane = column (vocab row of W); 16 consecutive f32 -> bf16.
// B k order used here (same as A): lanes 0-15 hold K=0..15, lanes 16-31 hold K=16..31
// (caller offsets ptr by kh = (lane>>4)*16).
__device__ __forceinline__ v16bf load_b_frag(const float* __restrict__ p) {
  v16bf b;
  const float4* q = (const float4*)p;
#pragma unroll
  for (int j = 0; j < 4; ++j) {
    float4 f = q[j];
    b[4 * j + 0] = (__bf16)f.x;
    b[4 * j + 1] = (__bf16)f.y;
    b[4 * j + 2] = (__bf16)f.z;
    b[4 * j + 3] = (__bf16)f.w;
  }
  return b;
}

// ---- 32x256 logits tile. acc[mt][nt] = 16x16 f32 accumulator.
// A (32 tokens x 128 K) staged in LDS as bf16 once per KSTEP; each staged
// chunk feeds 4 unrolled K=32 WMMA sub-steps x 2 M-tiles x 2 N-tiles.
__device__ __forceinline__ void compute_tile(const float* __restrict__ X,
                                             const float* __restrict__ W,
                                             int K, int tok0, int vbase,
                                             __bf16* ldsA, v8f acc[2][2]) {
  const int tid  = threadIdx.x;
  const int lane = tid & 31;
  const int wave = tid >> 5;
  const int m    = lane & 15;
  const int klo  = (lane >> 4) * 16;   // A frag: lanes 16-31 hold K=16..31
  const int kh   = (lane >> 4) * 16;   // B frag: lanes 16-31 hold K=16..31
  const int vcol = vbase + wave * 32 + (lane & 15);

  for (int k0 = 0; k0 < K; k0 += KSTEP) {
    __syncthreads();
    // stage 32x128 f32 -> bf16 LDS: 1024 float4 chunks, 4 per thread
#pragma unroll
    for (int t = 0; t < 4; ++t) {
      int c  = tid + t * THREADS;      // 0..1023
      int am = c >> 5;                 // row (32 float4 chunks per row)
      int kc = (c & 31) * 4;
      float4 f = *(const float4*)(X + (size_t)(tok0 + am) * K + k0 + kc);
      v4bf d;
      d[0] = (__bf16)f.x; d[1] = (__bf16)f.y;
      d[2] = (__bf16)f.z; d[3] = (__bf16)f.w;
      *(v4bf*)(ldsA + am * LDSTR + kc) = d;    // ds_store_b64, aligned
    }
    __syncthreads();

#pragma unroll
    for (int kk = 0; kk < KSTEP; kk += 32) {
      v16bf a[2];
#pragma unroll
      for (int mt = 0; mt < 2; ++mt) {
        const __bf16* ap = ldsA + (size_t)(mt * 16 + m) * LDSTR + kk + klo;
        v8bf alo = *(const v8bf*)ap;             // K = klo..klo+7
        v8bf ahi = *(const v8bf*)(ap + 8);       // K = klo+8..klo+15
#pragma unroll
        for (int e = 0; e < 8; ++e) { a[mt][e] = alo[e]; a[mt][8 + e] = ahi[e]; }
      }
#pragma unroll
      for (int nt = 0; nt < 2; ++nt) {
        const float* wrow = W + (size_t)(vcol + nt * 16) * K + k0 + kk + kh;
        if (kk == 0 && k0 + KSTEP < K)
          __builtin_prefetch(wrow + KSTEP, 0, 1);  // global_prefetch_b8
        v16bf b = load_b_frag(wrow);
#pragma unroll
        for (int mt = 0; mt < 2; ++mt) {
          acc[mt][nt] = __builtin_amdgcn_wmma_f32_16x16x32_bf16(
              false, a[mt], false, b, (short)0, acc[mt][nt], false, false);
        }
      }
    }
  }
}

// ---- per-token LSE over this 256-wide vocab chunk.
// C layout: element i of lane (half h) -> token mt*16 + i + 8h, col lane&15.
__device__ __forceinline__ void chunk_stats(const v8f acc[2][2], float* red,
                                            float* rowred,
                                            float* __restrict__ gout,
                                            int tok0, int ch) {
  const int tid  = threadIdx.x;
  const int lane = tid & 31;
  const int wave = tid >> 5;
  const int h    = lane >> 4;

  float mx[2][8];
#pragma unroll
  for (int mt = 0; mt < 2; ++mt)
#pragma unroll
    for (int i = 0; i < 8; ++i)
      mx[mt][i] = fmaxf(acc[mt][0][i], acc[mt][1][i]);
#pragma unroll
  for (int s = 1; s < 16; s <<= 1)
#pragma unroll
    for (int mt = 0; mt < 2; ++mt)
#pragma unroll
      for (int i = 0; i < 8; ++i)
        mx[mt][i] = fmaxf(mx[mt][i], __shfl_xor(mx[mt][i], s, 32));

  __syncthreads();
  if ((lane & 15) == 0) {
#pragma unroll
    for (int mt = 0; mt < 2; ++mt)
#pragma unroll
      for (int i = 0; i < 8; ++i)
        red[wave * TOKB + mt * 16 + h * 8 + i] = mx[mt][i];
  }
  __syncthreads();
  if (tid < TOKB) {
    float r = red[tid];
#pragma unroll
    for (int w = 1; w < WAVES; ++w) r = fmaxf(r, red[w * TOKB + tid]);
    rowred[tid] = r;
  }
  __syncthreads();

  float sm[2][8];
#pragma unroll
  for (int mt = 0; mt < 2; ++mt)
#pragma unroll
    for (int i = 0; i < 8; ++i) {
      float rm = rowred[mt * 16 + h * 8 + i];
      sm[mt][i] = __expf(acc[mt][0][i] - rm) + __expf(acc[mt][1][i] - rm);
    }
#pragma unroll
  for (int s = 1; s < 16; s <<= 1)
#pragma unroll
    for (int mt = 0; mt < 2; ++mt)
#pragma unroll
      for (int i = 0; i < 8; ++i)
        sm[mt][i] += __shfl_xor(sm[mt][i], s, 32);

  __syncthreads();
  if ((lane & 15) == 0) {
#pragma unroll
    for (int mt = 0; mt < 2; ++mt)
#pragma unroll
      for (int i = 0; i < 8; ++i)
        red[wave * TOKB + mt * 16 + h * 8 + i] = sm[mt][i];
  }
  __syncthreads();
  if (tid < TOKB) {
    float s = 0.f;
#pragma unroll
    for (int w = 0; w < WAVES; ++w) s += red[w * TOKB + tid];
    gout[(size_t)(tok0 + tid) * NCHUNKS + ch] = rowred[tid] + __logf(s);
  }
}

// ================= pass 1: chunk-level log-sum-exp ==================
__global__ void __launch_bounds__(THREADS)
jsd_pass1(const float* __restrict__ Xs, const float* __restrict__ Xt,
          const float* __restrict__ Ws, const float* __restrict__ Wt,
          float* __restrict__ lse_chunk_s, float* __restrict__ lse_chunk_t) {
  __shared__ __align__(16) __bf16 ldsA[TOKB * LDSTR];
  __shared__ float red[WAVES * TOKB];
  __shared__ float rowred[TOKB];

  const int tok0  = blockIdx.x * TOKB;
  const int ch    = blockIdx.y;
  const int vbase = ch * VCHUNK;

  v8f acc[2][2];
#pragma unroll
  for (int mt = 0; mt < 2; ++mt)
#pragma unroll
    for (int nt = 0; nt < 2; ++nt)
#pragma unroll
      for (int i = 0; i < 8; ++i) acc[mt][nt][i] = 0.f;
  compute_tile(Xs, Ws, HS, tok0, vbase, ldsA, acc);
  chunk_stats(acc, red, rowred, lse_chunk_s, tok0, ch);

#pragma unroll
  for (int mt = 0; mt < 2; ++mt)
#pragma unroll
    for (int nt = 0; nt < 2; ++nt)
#pragma unroll
      for (int i = 0; i < 8; ++i) acc[mt][nt][i] = 0.f;
  compute_tile(Xt, Wt, HT, tok0, vbase, ldsA, acc);
  chunk_stats(acc, red, rowred, lse_chunk_t, tok0, ch);
}

// ============ combine chunk LSEs into per-token LSE =================
__global__ void __launch_bounds__(THREADS)
lse_combine(const float* __restrict__ chunk, float* __restrict__ tok) {
  int t = blockIdx.x * blockDim.x + threadIdx.x;
  if (t >= BT) return;
  const float* row = chunk + (size_t)t * NCHUNKS;
  float mx = -3.4e38f;
  for (int c = 0; c < NCHUNKS; ++c) mx = fmaxf(mx, row[c]);
  float s = 0.f;
  for (int c = 0; c < NCHUNKS; ++c) s += __expf(row[c] - mx);
  tok[t] = mx + __logf(s);
}

// ======== pass 2: recompute logits, accumulate JSD terms ============
__global__ void __launch_bounds__(THREADS)
jsd_pass2(const float* __restrict__ Xs, const float* __restrict__ Xt,
          const float* __restrict__ Ws, const float* __restrict__ Wt,
          const float* __restrict__ lse_tok_s,
          const float* __restrict__ lse_tok_t,
          float* __restrict__ partial) {
  __shared__ __align__(16) __bf16 ldsA[TOKB * LDSTR];
  __shared__ float lsS[TOKB], lsT[TOKB];
  __shared__ float wred[WAVES];

  const int tid   = threadIdx.x;
  const int lane  = tid & 31;
  const int wave  = tid >> 5;
  const int h     = lane >> 4;
  const int tok0  = blockIdx.x * TOKB;
  const int vbase = blockIdx.y * VCHUNK;

  if (tid < TOKB) {
    lsS[tid] = lse_tok_s[tok0 + tid];
    lsT[tid] = lse_tok_t[tok0 + tid];
  }
  // barriers inside compute_tile make lsS/lsT visible before use below

  v8f accS[2][2], accT[2][2];
#pragma unroll
  for (int mt = 0; mt < 2; ++mt)
#pragma unroll
    for (int nt = 0; nt < 2; ++nt)
#pragma unroll
      for (int i = 0; i < 8; ++i) { accS[mt][nt][i] = 0.f; accT[mt][nt][i] = 0.f; }
  compute_tile(Xs, Ws, HS, tok0, vbase, ldsA, accS);
  compute_tile(Xt, Wt, HT, tok0, vbase, ldsA, accT);

  float tot = 0.f;
#pragma unroll
  for (int mt = 0; mt < 2; ++mt) {
#pragma unroll
    for (int nt = 0; nt < 2; ++nt) {
#pragma unroll
      for (int i = 0; i < 8; ++i) {
        int t = mt * 16 + h * 8 + i;
        float lq = accS[mt][nt][i] - lsS[t];
        float lp = accT[mt][nt][i] - lsT[t];
        float q  = __expf(lq);
        float p  = __expf(lp);
        float mm = 0.5f * (p + q);                 // BETA = 0.5 mixture
        float lm = __logf(fmaxf(mm, 1e-38f));
        tot += 0.5f * (p * (lp - lm) + q * (lq - lm));
      }
    }
  }
#pragma unroll
  for (int s = 1; s < 32; s <<= 1) tot += __shfl_xor(tot, s, 32);
  if (lane == 0) wred[wave] = tot;
  __syncthreads();
  if (tid == 0) {
    float s = 0.f;
#pragma unroll
    for (int w = 0; w < WAVES; ++w) s += wred[w];
    partial[(size_t)blockIdx.y * TBLKS + blockIdx.x] = s;
  }
}

// ====== deterministic fixed-order final reduction to the scalar =====
__global__ void __launch_bounds__(THREADS)
final_reduce(const float* __restrict__ partial, float* __restrict__ out) {
  __shared__ float buf[THREADS];
  float s = 0.f;
  for (int i = threadIdx.x; i < TBLKS * NCHUNKS; i += THREADS) s += partial[i];
  buf[threadIdx.x] = s;
  __syncthreads();
  for (int st = THREADS / 2; st > 0; st >>= 1) {
    if (threadIdx.x < st) buf[threadIdx.x] += buf[threadIdx.x + st];
    __syncthreads();
  }
  if (threadIdx.x == 0) out[0] = buf[0] * (1.0f / BT);
}

extern "C" void kernel_launch(void* const* d_in, const int* in_sizes, int n_in,
                              void* d_out, int out_size, void* d_ws, size_t ws_size,
                              hipStream_t stream) {
  const float* Xs = (const float*)d_in[0];  // [BT, HS]
  const float* Xt = (const float*)d_in[1];  // [BT, HT]
  const float* Ws = (const float*)d_in[2];  // [V, HS]
  const float* Wt = (const float*)d_in[3];  // [V, HT]

  float* ws          = (float*)d_ws;
  float* lse_chunk_s = ws;                                   // BT*NCHUNKS
  float* lse_chunk_t = lse_chunk_s + (size_t)BT * NCHUNKS;   // BT*NCHUNKS
  float* lse_tok_s   = lse_chunk_t + (size_t)BT * NCHUNKS;   // BT
  float* lse_tok_t   = lse_tok_s + BT;                       // BT
  float* partial     = lse_tok_t + BT;                       // TBLKS*NCHUNKS

  dim3 grid(TBLKS, NCHUNKS);  // x (token blocks) fastest -> shared W chunk in L2
  jsd_pass1<<<grid, THREADS, 0, stream>>>(Xs, Xt, Ws, Wt, lse_chunk_s, lse_chunk_t);
  lse_combine<<<BT / THREADS, THREADS, 0, stream>>>(lse_chunk_s, lse_tok_s);
  lse_combine<<<BT / THREADS, THREADS, 0, stream>>>(lse_chunk_t, lse_tok_t);
  jsd_pass2<<<grid, THREADS, 0, stream>>>(Xs, Xt, Ws, Wt, lse_tok_s, lse_tok_t, partial);
  final_reduce<<<1, THREADS, 0, stream>>>(partial, (float*)d_out);
}
